// Self_Attention_72258529787946
// MI455X (gfx1250) — matrix operand order repaired and checked by reading, not compiled, not run on hardware
//
#include <hip/hip_runtime.h>
#include <hip/hip_bf16.h>
#include <math.h>

// ---------------------------------------------------------------------------
// Self-attention for MI455X (gfx1250), bf16 WMMA with f32 accumulation.
//   B=4, S=4096, D=DOUT=1024
// Pipeline: convert(x)->bf16, convert(W)->bf16 transposed,
//           QKV GEMM (wmma), flash attention (wmma + online softmax).
// Round-2 changes: Q tile staged in LDS (ds_load_b128 A-fragments),
// fully parallel softmax reduction, global_prefetch of next K chunk,
// vectorized f32->bf16 conversion.
// ---------------------------------------------------------------------------

#define BATCH 4
#define SEQ   4096
#define DIM   1024   // D == DOUT
#define MTOT  (BATCH * SEQ)      // 16384 rows for QKV GEMM

typedef __attribute__((ext_vector_type(16))) __bf16 v16bf;
typedef __attribute__((ext_vector_type(8)))  float  v8f;

union FragBF {
  uint4 u[2];
  v16bf v;
};

// A-matrix 16x32 bf16 fragment (ISA 7.12.2): lane L holds row M=L&15.
// Lanes 0-15: K = d+0..7 (u[0]) and d+16..23 (u[1]); lanes 16-31: K = d+8..15
// and d+24..31.  Caller passes p = row_base + d + ((lane>>4)&1)*8.
__device__ __forceinline__ v16bf load_frag_a(const __bf16* p) {
  FragBF f;
  f.u[0] = *(const uint4*)(p);
  f.u[1] = *(const uint4*)(p + 16);
  return f.v;
}

// B-matrix 32x16 bf16 fragment: lane L holds column N=L&15; lanes 0-15 hold
// K = d+0..7 and d+16..23, lanes 16-31 hold K = d+8..15 and d+24..31
// (same K order as the A fragment).  Caller passes
// p = col_base + d + ((lane>>4)&1)*8.
__device__ __forceinline__ v16bf load_frag_b(const __bf16* p) {
  FragBF f;
  f.u[0] = *(const uint4*)(p);
  f.u[1] = *(const uint4*)(p + 16);
  return f.v;
}

// ---------------------------------------------------------------------------
// Kernel 1: x f32 -> bf16, 8 elements/thread (2x float4 in, 1x uint4 out).
__global__ void convert_x(const float* __restrict__ x,
                          __bf16* __restrict__ xb) {
  int i8 = (blockIdx.x * blockDim.x + threadIdx.x) * 8;
  float4 a = *(const float4*)(x + i8);
  float4 b = *(const float4*)(x + i8 + 4);
  union { __bf16 h[8]; uint4 u; } o;
  o.h[0] = (__bf16)a.x; o.h[1] = (__bf16)a.y;
  o.h[2] = (__bf16)a.z; o.h[3] = (__bf16)a.w;
  o.h[4] = (__bf16)b.x; o.h[5] = (__bf16)b.y;
  o.h[6] = (__bf16)b.z; o.h[7] = (__bf16)b.w;
  *(uint4*)(xb + i8) = o.u;
}

// Kernel 2: W [3][D][DOUT] f32 -> Wt [3][DOUT][D] bf16 (transposed so WMMA
// B-fragments are contiguous 128-bit loads).  Output-linear indexing keeps
// the bf16 stores coalesced.
__global__ void convert_w(const float* __restrict__ w, __bf16* __restrict__ wt) {
  int i = blockIdx.x * blockDim.x + threadIdx.x;       // over 3*D*DOUT
  int mat = i >> 20;               // / (1024*1024)
  int r   = i & ((1 << 20) - 1);
  int e   = r >> 10;               // output row  (DOUT index)
  int d   = r & 1023;              // output col  (D index)
  wt[i] = (__bf16)w[(size_t)mat * DIM * DIM + (size_t)d * DIM + e];
}

// ---------------------------------------------------------------------------
// Kernel 3: QKV projection.  C[mat] = Xb(16384x1024) * W[mat](1024x1024).
// Block = 128 threads = 4 waves; each wave computes a 16x64 tile (4 wmma
// accumulators), A-fragment reused across the 4 N-tiles.
// mat 0 -> Kb row-major, mat 1 -> Qb row-major, mat 2 -> Vt transposed
// ([B][DOUT][S]) so the attention P*V B-fragments are contiguous.
__global__ void gemm_qkv(const __bf16* __restrict__ xb,
                         const __bf16* __restrict__ wt,
                         __bf16* __restrict__ qb,
                         __bf16* __restrict__ kb,
                         __bf16* __restrict__ vt) {
  const int tid  = threadIdx.x;
  const int wave = tid >> 5;
  const int lane = tid & 31;
  const int rl    = lane & 15;
  const int hi    = (lane >> 4) & 1;
  const int rbase = hi << 3;          // C/D row offset for this lane half
  const int koffA = hi * 8;
  const int koffB = hi * 8;

  const int mat = blockIdx.z;
  const int m0  = (blockIdx.y * 4 + wave) * 16;
  const int n0  = blockIdx.x * 64;

  const __bf16* arow = xb + (size_t)(m0 + rl) * DIM;
  const __bf16* wmat = wt + (size_t)mat * DIM * DIM;

  v8f acc[4];
#pragma unroll
  for (int j = 0; j < 4; ++j) acc[j] = (v8f)(0.0f);

#pragma unroll 4
  for (int d = 0; d < DIM; d += 32) {
    v16bf a = load_frag_a(arow + d + koffA);
#pragma unroll
    for (int j = 0; j < 4; ++j) {
      const __bf16* bcol = wmat + (size_t)(n0 + 16 * j + rl) * DIM;
      v16bf b = load_frag_b(bcol + d + koffB);
      acc[j] = __builtin_amdgcn_wmma_f32_16x16x32_bf16(
          false, a, false, b, (short)0, acc[j], false, false);
    }
  }

#pragma unroll
  for (int j = 0; j < 4; ++j) {
#pragma unroll
    for (int r = 0; r < 8; ++r) {
      int row = m0 + rbase + r;
      int col = n0 + 16 * j + rl;
      __bf16 val = (__bf16)acc[j][r];
      if (mat == 0) {
        kb[(size_t)row * DIM + col] = val;
      } else if (mat == 1) {
        qb[(size_t)row * DIM + col] = val;
      } else {
        int b = row >> 12;            // row / SEQ
        int s = row & (SEQ - 1);
        vt[((size_t)b * DIM + col) * SEQ + s] = val;
      }
    }
  }
}

// ---------------------------------------------------------------------------
// Kernel 4: flash attention.  One block (256 thr = 8 waves) per 16-query
// tile.  Q tile (16x1024 bf16, 32KB) staged once in LDS; wave w owns output
// columns [w*128, w*128+128) as 8 f32 wmma accumulators.  Per 128-key chunk:
// wave w computes score tile (16q x 16k) -> LDS, parallel online softmax
// (256-thread two-stage LDS reduction) -> P bf16 in LDS, accumulator rescale
// by alpha, then O += P * V.  Next chunk's K rows are prefetched.
__global__ void __launch_bounds__(256)
attention(const __bf16* __restrict__ qb, const __bf16* __restrict__ kb,
          const __bf16* __restrict__ vt, float* __restrict__ out) {
  __shared__ __bf16 sQ[16][DIM];       // 32 KB staged Q tile
  __shared__ float  sS[16][128];       //  8 KB score chunk
  __shared__ __bf16 sP[16][128];       //  4 KB probabilities
  __shared__ float  sRed[16][16];      //  1 KB reduction scratch
  __shared__ float  sM[16], sL[16], sA[16];

  const int tid  = threadIdx.x;
  const int wave = tid >> 5;
  const int lane = tid & 31;
  const int rl    = lane & 15;
  const int hi    = (lane >> 4) & 1;
  const int rbase = hi << 3;
  const int koffA = hi * 8;
  const int koffB = hi * 8;

  const int q0 = blockIdx.x * 16;      // global query row (flattened B*S)
  const int b  = q0 >> 12;             // q0 / SEQ
  const float scale = 0.03125f;        // 1/sqrt(1024)

  const __bf16* kbat = kb + (size_t)b * SEQ * DIM;
  const __bf16* vbat = vt + (size_t)b * DIM * SEQ;     // [DOUT][S]

  // ---- stage Q tile (16 consecutive rows are contiguous): 32KB flat copy
  {
    const uint4* src = (const uint4*)(qb + (size_t)q0 * DIM);
    uint4* dst = (uint4*)(&sQ[0][0]);
#pragma unroll
    for (int it = 0; it < 8; ++it) dst[tid + it * 256] = src[tid + it * 256];
  }

  v8f acc[8];
#pragma unroll
  for (int t = 0; t < 8; ++t) acc[t] = (v8f)(0.0f);

  if (tid < 16) { sM[tid] = -1.0e30f; sL[tid] = 0.0f; }
  __syncthreads();

  const int srow = tid >> 4;           // softmax row for this thread
  const int sseg = tid & 15;           // 8-column segment within row

  for (int kc = 0; kc < SEQ; kc += 128) {
    // ---- prefetch next chunk's K rows (256B stride) ----
    if (kc + 128 < SEQ) {
      const __bf16* knext = kbat + (size_t)(kc + 128 + 16 * wave + rl) * DIM;
#pragma unroll
      for (int pf = 0; pf < DIM; pf += 128)
        __builtin_prefetch(knext + pf, 0, 1);
    }

    // ---- scores: wave w -> key columns kc+16w .. kc+16w+15 ----
    v8f sc = (v8f)(0.0f);
    const __bf16* kcol = kbat + (size_t)(kc + 16 * wave + rl) * DIM;
#pragma unroll 4
    for (int d = 0; d < DIM; d += 32) {
      v16bf a  = load_frag_a(&sQ[rl][d + koffA]);        // ds_load_b128
      v16bf bf = load_frag_b(kcol + d + koffB);
      sc = __builtin_amdgcn_wmma_f32_16x16x32_bf16(
          false, a, false, bf, (short)0, sc, false, false);
    }
    {
      int col = 16 * wave + rl;
#pragma unroll
      for (int r = 0; r < 8; ++r) sS[rbase + r][col] = sc[r];
    }
    __syncthreads();

    // ---- softmax phase A: per-thread partial max over 8 columns ----
    {
      float pm = -1.0e30f;
#pragma unroll
      for (int c = 0; c < 8; ++c)
        pm = fmaxf(pm, sS[srow][sseg * 8 + c] * scale);
      sRed[srow][sseg] = pm;
    }
    __syncthreads();

    // ---- softmax phase B: finalize row max + rescale factor ----
    if (tid < 16) {
      float mx = sM[tid];
#pragma unroll
      for (int s = 0; s < 16; ++s) mx = fmaxf(mx, sRed[tid][s]);
      sA[tid] = __expf(sM[tid] - mx);
      sM[tid] = mx;
    }
    __syncthreads();

    // ---- softmax phase C: exp + partial sums, write P (bf16) ----
    {
      float mx = sM[srow];
      float sum = 0.0f;
#pragma unroll
      for (int c = 0; c < 8; ++c) {
        float e = __expf(sS[srow][sseg * 8 + c] * scale - mx);
        sP[srow][sseg * 8 + c] = (__bf16)e;
        sum += e;
      }
      sRed[srow][sseg] = sum;
    }
    __syncthreads();

    // ---- softmax phase D: accumulate denominator ----
    if (tid < 16) {
      float sum = 0.0f;
#pragma unroll
      for (int s = 0; s < 16; ++s) sum += sRed[tid][s];
      sL[tid] = sL[tid] * sA[tid] + sum;
    }

    // ---- rescale accumulators by alpha, then O += P * V ----
    float alphas[8];
#pragma unroll
    for (int r = 0; r < 8; ++r) alphas[r] = sA[rbase + r];
#pragma unroll
    for (int t = 0; t < 8; ++t)
#pragma unroll
      for (int r = 0; r < 8; ++r) acc[t][r] *= alphas[r];

#pragma unroll
    for (int ks = 0; ks < 4; ++ks) {               // 32 keys per wmma step
      v16bf a = load_frag_a(&sP[rl][ks * 32 + koffA]);
#pragma unroll
      for (int t = 0; t < 8; ++t) {
        int ncol = wave * 128 + t * 16 + rl;
        const __bf16* vcol = vbat + (size_t)ncol * SEQ + kc + ks * 32 + koffB;
        v16bf bf = load_frag_b(vcol);
        acc[t] = __builtin_amdgcn_wmma_f32_16x16x32_bf16(
            false, a, false, bf, (short)0, acc[t], false, false);
      }
    }
    __syncthreads();
  }

  // ---- epilogue: divide by softmax denominator, store f32 ----
  float linv[8];
#pragma unroll
  for (int r = 0; r < 8; ++r) linv[r] = 1.0f / sL[rbase + r];
#pragma unroll
  for (int t = 0; t < 8; ++t) {
#pragma unroll
    for (int r = 0; r < 8; ++r) {
      int row = q0 + rbase + r;
      int col = wave * 128 + t * 16 + rl;
      out[(size_t)row * DIM + col] = acc[t][r] * linv[r];
    }
  }
}

// ---------------------------------------------------------------------------
extern "C" void kernel_launch(void* const* d_in, const int* in_sizes, int n_in,
                              void* d_out, int out_size, void* d_ws,
                              size_t ws_size, hipStream_t stream) {
  const float* x = (const float*)d_in[0];      // [4,4096,1024] f32
  const float* w = (const float*)d_in[1];      // [3,1024,1024] f32
  float* out = (float*)d_out;                  // [4,4096,1024] f32

  // Workspace layout (bytes):
  //   Xb : 0          .. 32MB   bf16 [16384][1024]
  //   Wt : 32MB       .. +6MB   bf16 [3][1024][1024] (transposed)
  //   Qb : 38MB       .. +32MB  bf16 [16384][1024]
  //   Kb : 70MB       .. +32MB  bf16 [16384][1024]
  //   Vt : 102MB      .. +32MB  bf16 [4][1024][4096] (transposed)
  char* ws = (char*)d_ws;
  const size_t SZ_XB = (size_t)MTOT * DIM * 2;       // 32 MB
  const size_t SZ_WT = (size_t)3 * DIM * DIM * 2;    //  6 MB
  __bf16* Xb = (__bf16*)(ws);
  __bf16* Wt = (__bf16*)(ws + SZ_XB);
  __bf16* Qb = (__bf16*)(ws + SZ_XB + SZ_WT);
  __bf16* Kb = (__bf16*)(ws + SZ_XB + SZ_WT + SZ_XB);
  __bf16* Vt = (__bf16*)(ws + SZ_XB + SZ_WT + 2 * SZ_XB);

  const int nX = MTOT * DIM;        // 16,777,216
  const int nW = 3 * DIM * DIM;     //  3,145,728

  convert_x<<<nX / (256 * 8), 256, 0, stream>>>(x, Xb);
  convert_w<<<nW / 256, 256, 0, stream>>>(w, Wt);

  // 16 N-tiles (64 cols), 256 M-blocks (4 waves * 16 rows), 3 matrices
  gemm_qkv<<<dim3(16, 256, 3), 128, 0, stream>>>(Xb, Wt, Qb, Kb, Vt);

  // one block per 16-query tile: (B*S)/16 = 1024 blocks
  attention<<<MTOT / 16, 256, 0, stream>>>(Qb, Kb, Vt, out);
}
